// MultiheadAttention_7155415515178
// MI455X (gfx1250) — matrix operand order repaired and checked by reading, not compiled, not run on hardware
//
#include <hip/hip_runtime.h>
#include <stdint.h>

#define B_  2
#define S_  2048
#define D_  1024
#define H_  16
#define HD_ 64

typedef _Float16 v4h  __attribute__((ext_vector_type(4)));
typedef _Float16 v8h  __attribute__((ext_vector_type(8)));
typedef _Float16 v16h __attribute__((ext_vector_type(16)));
typedef float    v4f  __attribute__((ext_vector_type(4)));
typedef float    v8f  __attribute__((ext_vector_type(8)));
typedef unsigned int u32x4 __attribute__((ext_vector_type(4)));
typedef int          i32x4 __attribute__((ext_vector_type(4)));
typedef int          i32x8 __attribute__((ext_vector_type(8)));

static __device__ __forceinline__ v8h ld8(const _Float16* p) { return *(const v8h*)p; }

static __device__ __forceinline__ v16h cat8(v8h lo, v8h hi) {
  return __builtin_shufflevector(lo, hi, 0,1,2,3,4,5,6,7,8,9,10,11,12,13,14,15);
}

// D = A(16x32,f16) * B(32x16,f16) + C(f32)
static __device__ __forceinline__ v8f wmma_f16(v16h a, v16h b, v8f c) {
  return __builtin_amdgcn_wmma_f32_16x16x32_f16(false, a, false, b, (short)0, c,
                                                false, false);
}

// raw LDS byte offset of a __shared__ object (generic -> AS(3) -> int)
static __device__ __forceinline__ uint32_t lds_off(const void* p) {
  return (uint32_t)(uintptr_t)(__attribute__((address_space(3))) const void*)p;
}

// ------------------------------------------------------------------ TDM load
// 2D tile of f16 (data_size=2B): tile_w elems per row, tile_h rows, global row
// stride in elements; optional LDS pad insertion after each row.
// pad_icode: 0=2,1=4,2=8,3=16,4=32.. DWORDs before pad; pad_acode: n -> n+1 DWORDs.
static __device__ __forceinline__ void tdm_load_2d(uint32_t lds_addr,
                                                   uint64_t gaddr,
                                                   uint32_t tile_w, uint32_t tile_h,
                                                   uint32_t row_stride,
                                                   uint32_t pad_icode,
                                                   uint32_t pad_acode) {
  u32x4 g0;
  g0[0] = 1u;                                           // count=1 (valid user D#)
  g0[1] = lds_addr;                                     // lds_addr [63:32]
  g0[2] = (uint32_t)gaddr;                              // global_addr lo
  g0[3] = (uint32_t)((gaddr >> 32) & 0x01FFFFFFu)       // global_addr hi [120:96]
        | (2u << 30);                                   // type=2 ("image")
  i32x8 g1;
  g1[0] = (int)((1u << 16)                              // data_size = 2 bytes
        | (1u << 20)                                    // pad_enable
        | (pad_icode << 22) | (pad_acode << 25));
  g1[1] = (int)((tile_w & 0xFFFFu) << 16);              // tensor_dim0 lo (=tile_w)
  g1[2] = (int)(((tile_w >> 16) & 0xFFFFu)              // tensor_dim0 hi
        | ((tile_h & 0xFFFFu) << 16));                  // tensor_dim1 lo (=tile_h)
  g1[3] = (int)(((tile_h >> 16) & 0xFFFFu)              // tensor_dim1 hi
        | ((tile_w & 0xFFFFu) << 16));                  // tile_dim0
  g1[4] = (int)(tile_h & 0xFFFFu);                      // tile_dim1 (tile_dim2=0)
  g1[5] = (int)row_stride;                              // tensor_dim0_stride lo
  g1[6] = 0;                                            // stride hi / dim1_stride
  g1[7] = 0;
  const i32x4 gz = {0, 0, 0, 0};
#if __clang_major__ >= 23
  const i32x8 gz8 = {0, 0, 0, 0, 0, 0, 0, 0};
  __builtin_amdgcn_tensor_load_to_lds(g0, g1, gz, gz, gz8, 0);
#else
  __builtin_amdgcn_tensor_load_to_lds(g0, g1, gz, gz, 0);
#endif
}

// ---------------------------------------------------------------- f32 -> f16
__global__ __launch_bounds__(256) void cvt_f32_f16(const float* __restrict__ in,
                                                   _Float16* __restrict__ out,
                                                   int n4) {
  int i = blockIdx.x * 256 + threadIdx.x;
  if (i < n4) {
    v4f x = *(const v4f*)(in + 4 * (size_t)i);
    v4h y = __builtin_convertvector(x, v4h);
    *(v4h*)(out + 4 * (size_t)i) = y;
  }
}

// -------------------------------------------------- C = A * W^T + bias GEMM
// A: M x K f16 row-major.  W: N x K f16 row-major (torch Linear weight).
// 128x128 tile, 8 waves x (32x64).  A/B tiles DMA'd by the TDM (wave 0) into
// double-buffered LDS with hardware pad insertion; compute overlaps the DMA.
template <bool F32OUT>
__global__ __launch_bounds__(256) void wmma_gemm_bias(
    const _Float16* __restrict__ A, const _Float16* __restrict__ W,
    const float* __restrict__ bias, void* __restrict__ Cout,
    int M, int N, int K) {
  __shared__ _Float16 As[2][128][40];   // 32 data + 8 pad halves per row
  __shared__ _Float16 Bs[2][128][40];

  const int tid  = threadIdx.x;
  const int lane = tid & 31;
  const int wave = tid >> 5;
  const int wm   = (wave >> 1) * 32;
  const int wn   = (wave & 1) * 64;
  const int m0   = blockIdx.y * 128;
  const int n0   = blockIdx.x * 128;
  const int ln   = lane & 15;
  const int kg   = lane >> 4;

  const v8f vz = {0.f, 0.f, 0.f, 0.f, 0.f, 0.f, 0.f, 0.f};
  v8f acc[2][4];
#pragma unroll
  for (int i = 0; i < 2; i++)
#pragma unroll
    for (int j = 0; j < 4; j++) acc[i][j] = vz;

  const uint64_t abase = (uint64_t)(uintptr_t)A + (uint64_t)m0 * K * 2;
  const uint64_t bbase = (uint64_t)(uintptr_t)W + (uint64_t)n0 * K * 2;
  const bool tdm_wave = (wave == 0);

  if (tdm_wave) {                       // prologue: fetch tile 0 into buffer 0
    tdm_load_2d(lds_off(&As[0][0][0]), abase, 32, 128, (uint32_t)K, 3, 3);
    tdm_load_2d(lds_off(&Bs[0][0][0]), bbase, 32, 128, (uint32_t)K, 3, 3);
  }

  for (int k0 = 0; k0 < K; k0 += 32) {
    const int buf = (k0 >> 5) & 1;
    __builtin_amdgcn_s_wait_tensorcnt(0);   // wave0: current tile landed (others: cnt==0)
    __syncthreads();                        // everyone done with buf^1, buf visible
    if (tdm_wave && (k0 + 32) < K) {        // prefetch next tile into other buffer
      tdm_load_2d(lds_off(&As[buf ^ 1][0][0]), abase + (uint64_t)(k0 + 32) * 2,
                  32, 128, (uint32_t)K, 3, 3);
      tdm_load_2d(lds_off(&Bs[buf ^ 1][0][0]), bbase + (uint64_t)(k0 + 32) * 2,
                  32, 128, (uint32_t)K, 3, 3);
    }

    v16h af[2], bf[4];
#pragma unroll
    for (int i = 0; i < 2; i++) {        // A layout: half-wave K split, contiguous
      const _Float16* r = &As[buf][wm + i * 16 + ln][0];
      af[i] = cat8(ld8(r + 16 * kg), ld8(r + 16 * kg + 8));
    }
#pragma unroll
    for (int j = 0; j < 4; j++) {        // B layout: half-wave K split, contiguous
      const _Float16* r = &Bs[buf][wn + j * 16 + ln][0];
      bf[j] = cat8(ld8(r + 16 * kg), ld8(r + 16 * kg + 8));
    }
#pragma unroll
    for (int i = 0; i < 2; i++)
#pragma unroll
      for (int j = 0; j < 4; j++)
        acc[i][j] = wmma_f16(af[i], bf[j], acc[i][j]);
  }

#pragma unroll
  for (int j = 0; j < 4; j++) {
    const int col = n0 + wn + j * 16 + ln;
    const float bc = bias[col];
#pragma unroll
    for (int i = 0; i < 2; i++) {
      const int rbase = m0 + wm + i * 16 + 8 * kg;   // C layout rows
#pragma unroll
      for (int r = 0; r < 8; r++) {
        const float v = acc[i][j][r] + bc;
        const size_t idx = (size_t)(rbase + r) * N + col;
        if (F32OUT) ((float*)Cout)[idx] = v;
        else        ((_Float16*)Cout)[idx] = (_Float16)v;
      }
    }
  }
}

// ------------------------------------------------------- causal flash attn
// Q/K/V: per-head (S, 64) f16 (reshape of projected (B,S,D) buffer).
// Out: f16 (B, S, D) with col = h*64 + hd  (the transpose(0,2,1,3) merge).
// K tile arrives via TDM (padded rows); V is transposed into LDS by the waves.
__global__ __launch_bounds__(256) void flash_attn_causal(
    const _Float16* __restrict__ Q, const _Float16* __restrict__ Kh,
    const _Float16* __restrict__ V, _Float16* __restrict__ Out) {
  __shared__ _Float16 Ks[32][72];        // K block: 32 krows x 64 hd (+8 pad)
  __shared__ _Float16 Vt[64][40];        // V block transposed: 64 hd x 32 k (+pad)
  __shared__ _Float16 Ps[8][16][32];     // per-wave P tile (C-layout -> A-layout)

  const int tid  = threadIdx.x;
  const int lane = tid & 31;
  const int wave = tid >> 5;
  const int ln   = lane & 15;
  const int kg   = lane >> 4;
  const int bh   = blockIdx.y;           // b*H + h
  const int b    = bh >> 4;
  const int h    = bh & 15;
  const size_t hbase = (size_t)bh * S_ * HD_;
  const int q0   = blockIdx.x * 128 + wave * 16;

  // Q fragments (A layout): row q0+ln, hd chunks 0-31 / 32-63
  const _Float16* qrow = Q + hbase + (size_t)(q0 + ln) * HD_;
  v16h qf[2];
#pragma unroll
  for (int c = 0; c < 2; c++)
    qf[c] = cat8(ld8(qrow + c * 32 + 16 * kg), ld8(qrow + c * 32 + 16 * kg + 8));

  const v8f vz = {0.f, 0.f, 0.f, 0.f, 0.f, 0.f, 0.f, 0.f};
  float mi[8], li[8];
  v8f oacc[4];
#pragma unroll
  for (int r = 0; r < 8; r++) { mi[r] = -3.0e38f; li[r] = 0.f; }
#pragma unroll
  for (int t = 0; t < 4; t++) oacc[t] = vz;

  const int lrow = tid >> 3;             // cooperative V loader: 0..31 k row
  const int lcol = (tid & 7) * 8;        // 8 hd per thread

  const uint64_t kbytes = (uint64_t)(uintptr_t)Kh + (uint64_t)hbase * 2;
  const int kend = blockIdx.x * 128 + 128;
  for (int kb0 = 0; kb0 < kend; kb0 += 32) {
    __syncthreads();                     // previous iteration's LDS reads done
    if (wave == 0)                       // TDM: K tile 32x64 f16, 128B rows + pad
      tdm_load_2d(lds_off(&Ks[0][0]), kbytes + (uint64_t)kb0 * HD_ * 2,
                  64, 32, 64, 4, 3);
    {                                    // V transpose store (all waves)
      v8h vv = ld8(V + hbase + (size_t)(kb0 + lrow) * HD_ + lcol);
#pragma unroll
      for (int e = 0; e < 8; e++) Vt[lcol + e][lrow] = vv[e];
    }
    __builtin_amdgcn_s_wait_tensorcnt(0);
    __syncthreads();

    const bool active = (kb0 <= q0 + 15);   // wave has any unmasked column
    if (active) {
      // logits S = Q * K^T  (two 16-col tiles, contraction hd=64 in 2 chunks)
      v8f sc[2];
#pragma unroll
      for (int t = 0; t < 2; t++) {
        sc[t] = vz;
#pragma unroll
        for (int c = 0; c < 2; c++) {
          const _Float16* r = &Ks[t * 16 + ln][c * 32 + 16 * kg];
          v16h bf = cat8(ld8(r), ld8(r + 8));
          sc[t] = wmma_f16(qf[c], bf, sc[t]);
        }
      }
      // online softmax over C-layout rows (row = r + 8*kg across 16 lanes)
      float alpha[8];
#pragma unroll
      for (int r = 0; r < 8; r++) {
        const int qg = q0 + r + 8 * kg;
        float s0 = sc[0][r] * 0.125f;            // 1/sqrt(64)
        float s1 = sc[1][r] * 0.125f;
        if (kb0 + ln > qg)      s0 = -1.0e30f;   // causal mask
        if (kb0 + 16 + ln > qg) s1 = -1.0e30f;
        float mx = fmaxf(s0, s1);
#pragma unroll
        for (int o = 1; o < 16; o <<= 1) mx = fmaxf(mx, __shfl_xor(mx, o, 16));
        const float mn = fmaxf(mi[r], mx);
        alpha[r] = __expf(mi[r] - mn);
        const float p0 = __expf(s0 - mn);
        const float p1 = __expf(s1 - mn);
        float rs = p0 + p1;
#pragma unroll
        for (int o = 1; o < 16; o <<= 1) rs += __shfl_xor(rs, o, 16);
        li[r] = li[r] * alpha[r] + rs;
        mi[r] = mn;
        Ps[wave][r + 8 * kg][ln]      = (_Float16)p0;   // C-layout -> LDS
        Ps[wave][r + 8 * kg][16 + ln] = (_Float16)p1;
      }
#pragma unroll
      for (int t = 0; t < 4; t++)
#pragma unroll
        for (int r = 0; r < 8; r++) oacc[t][r] *= alpha[r];

      // reload P as an A-fragment (wave-private LDS patch; DS in-order per wave)
      const _Float16* pr = &Ps[wave][ln][0];
      v16h pf = cat8(ld8(pr + 16 * kg), ld8(pr + 16 * kg + 8));
#pragma unroll
      for (int t = 0; t < 4; t++) {
        const _Float16* vr = &Vt[t * 16 + ln][16 * kg];
        v16h vf = cat8(ld8(vr), ld8(vr + 8));
        oacc[t] = wmma_f16(pf, vf, oacc[t]);
      }
    }
  }

  // normalize and store in merged-head (B,S,D) layout for the output GEMM
  float inv[8];
#pragma unroll
  for (int r = 0; r < 8; r++) inv[r] = 1.0f / li[r];
#pragma unroll
  for (int t = 0; t < 4; t++) {
    const int col = h * 64 + t * 16 + ln;
#pragma unroll
    for (int r = 0; r < 8; r++) {
      const int s = q0 + r + 8 * kg;
      Out[((size_t)b * S_ + s) * D_ + col] = (_Float16)(oacc[t][r] * inv[r]);
    }
  }
}

// ---------------------------------------------------------------- launcher
extern "C" void kernel_launch(void* const* d_in, const int* in_sizes, int n_in,
                              void* d_out, int out_size, void* d_ws, size_t ws_size,
                              hipStream_t stream) {
  (void)in_sizes; (void)n_in; (void)out_size; (void)ws_size;
  const float* q  = (const float*)d_in[0];
  const float* k  = (const float*)d_in[1];
  const float* v  = (const float*)d_in[2];
  // d_in[3] = mask: causal, implemented analytically
  const float* Wq = (const float*)d_in[4];
  const float* bq = (const float*)d_in[5];
  const float* Wk = (const float*)d_in[6];
  const float* bk = (const float*)d_in[7];
  const float* Wv = (const float*)d_in[8];
  const float* bv = (const float*)d_in[9];
  const float* Wo = (const float*)d_in[10];
  const float* bo = (const float*)d_in[11];

  const size_t X = (size_t)B_ * S_ * D_;   // 4194304
  const size_t W = (size_t)D_ * D_;        // 1048576
  _Float16* ws  = (_Float16*)d_ws;         // total 7X + 4W halves = 64 MB
  _Float16* qin = ws;
  _Float16* kin = qin + X;
  _Float16* vin = kin + X;
  _Float16* wq  = vin + X;
  _Float16* wk  = wq + W;
  _Float16* wv  = wk + W;
  _Float16* wo  = wv + W;
  _Float16* qp  = wo + W;
  _Float16* kp  = qp + X;
  _Float16* vp  = kp + X;
  _Float16* ao  = vp + X;

  auto cvt = [&](const float* src, _Float16* dst, size_t n) {
    int n4 = (int)(n / 4);
    cvt_f32_f16<<<dim3((n4 + 255) / 256), dim3(256), 0, stream>>>(src, dst, n4);
  };
  cvt(q, qin, X);  cvt(k, kin, X);  cvt(v, vin, X);
  cvt(Wq, wq, W);  cvt(Wk, wk, W);  cvt(Wv, wv, W);  cvt(Wo, wo, W);

  dim3 gg(D_ / 128, (B_ * S_) / 128);   // (8, 32)
  wmma_gemm_bias<false><<<gg, 256, 0, stream>>>(qin, wq, bq, qp, B_ * S_, D_, D_);
  wmma_gemm_bias<false><<<gg, 256, 0, stream>>>(kin, wk, bk, kp, B_ * S_, D_, D_);
  wmma_gemm_bias<false><<<gg, 256, 0, stream>>>(vin, wv, bv, vp, B_ * S_, D_, D_);

  flash_attn_causal<<<dim3(S_ / 128, B_ * H_), 256, 0, stream>>>(qp, kp, vp, ao);

  wmma_gemm_bias<true><<<gg, 256, 0, stream>>>(ao, wo, bo, d_out, B_ * S_, D_, D_);
}
